// ROIDCN_9002251452666
// MI455X (gfx1250) — matrix operand order repaired and checked by reading, not compiled, not run on hardware
//
#include <hip/hip_runtime.h>
#include <hip/hip_bf16.h>

typedef __attribute__((ext_vector_type(16))) __bf16 v16bf;
typedef __attribute__((ext_vector_type(2)))  __bf16 v2bf;
typedef __attribute__((ext_vector_type(8)))  float  v8f;

#define C_IN   256
#define C_OUT  256
#define HH     128
#define WW     128
#define HW     (HH * WW)          // 16384
#define PP     HW
#define KTAPS  9
#define KDIM   (KTAPS * C_IN)     // 2304, ordered k_tap*256 + c
#define KCH    32                 // K per WMMA step
#define NCHUNK (KDIM / KCH)       // 72
#define NTILE  64                 // positions per workgroup (four 16-wide N tiles)
#define BROW   48                 // padded LDS B-row length in halves (96B, 16B aligned)

__device__ __forceinline__ unsigned short f2bf(float f) {
    unsigned int u = __builtin_bit_cast(unsigned int, f);
    u += 0x7FFFu + ((u >> 16) & 1u);      // round-to-nearest-even
    return (unsigned short)(u >> 16);
}

__device__ __forceinline__ unsigned int pack_bf16x2(float lo, float hi) {
    v2bf p;
    p.x = (__bf16)lo;                     // hardware RNE convert
    p.y = (__bf16)hi;
    return __builtin_bit_cast(unsigned int, p);
}

// ---- Kernel 1: repack weight fp32 (O, C, 3, 3) -> bf16 [o][k_tap*256 + c] ----
__global__ __launch_bounds__(256) void roidcn_repack_w(const float* __restrict__ w,
                                                       unsigned short* __restrict__ wbf) {
    int t = blockIdx.x * blockDim.x + threadIdx.x;
    if (t >= C_OUT * KDIM) return;
    int o    = t / KDIM;
    int kidx = t - o * KDIM;      // = k_tap*256 + c
    int ktap = kidx >> 8;
    int c    = kidx & 255;
    float v  = w[(size_t)o * (C_IN * KTAPS) + (size_t)c * KTAPS + ktap];
    wbf[t] = f2bf(v);
}

// ---- Kernel 2: fused bilinear im2col + bf16 WMMA GEMM + ReLU ----
__global__ __launch_bounds__(256) void roidcn_main(const float* __restrict__ x,
                                                   const float* __restrict__ bboxes,
                                                   const float* __restrict__ scores,
                                                   const unsigned short* __restrict__ wbf,
                                                   float* __restrict__ out) {
    __shared__ int            sIdx[NTILE * KTAPS][4];   // bilinear corner flat indices
    __shared__ float          sWgt[NTILE * KTAPS][4];   // bilinear corner weights (masked)
    __shared__ unsigned short sB[2][NTILE * BROW];      // double-buffered B fragments

    const int tid   = threadIdx.x;
    const int lane  = tid & 31;
    const int wave  = tid >> 5;               // 0..7
    const int b     = blockIdx.x >> 8;        // grid = 2 * 256 tiles
    const int pbase = (blockIdx.x & 255) * NTILE;

    // ---- sampling coordinates: 64 positions x 9 taps ----
    for (int t = tid; t < NTILE * KTAPS; t += 256) {
        const int p  = t / KTAPS;
        const int k  = t - p * KTAPS;
        const int pg = pbase + p;
        const float xx = (float)(k % 3 - 1);
        const float yy = (float)(k / 3 - 1);
        const float* bb = bboxes + ((size_t)b * PP + pg) * 4;
        const float x1 = bb[0], y1 = bb[1], x2 = bb[2], y2 = bb[3];
        (void)y1;
        const float sc = scores[(size_t)b * PP + pg];
        const float inv_s = 0.125f;                    // 1/stride, stride==8
        const float x_ctr = (x1 + x2) * 0.5f * inv_s;
        const float y_ctr = (y1 + y2) * 0.5f * inv_s;
        const float dw    = (x2 - x1) * inv_s * (1.0f / 3.0f);
        // reference: dh = (y2 - y2)/stride/K == 0 -> y_bbox = y_ctr
        const float xbb = dw * xx + x_ctr;
        const float ybb = y_ctr;
        const float xcv = (float)(pg % WW) + xx;
        const float ycv = (float)(pg / WW) + yy;
        const float px = xcv + (xbb - xcv) * sc;
        const float py = ycv + (ybb - ycv) * sc;
        const float fx0 = floorf(px), fy0 = floorf(py);
        const float wx = px - fx0, wy = py - fy0;
        const int ix0 = (int)fx0, iy0 = (int)fy0;
#pragma unroll
        for (int d = 0; d < 4; ++d) {
            const int dy = d >> 1, dx = d & 1;
            const int yi = iy0 + dy, xi = ix0 + dx;
            const bool valid = (yi >= 0) && (yi < HH) && (xi >= 0) && (xi < WW);
            const int yic = min(max(yi, 0), HH - 1);
            const int xic = min(max(xi, 0), WW - 1);
            const float wv = (dy ? wy : 1.0f - wy) * (dx ? wx : 1.0f - wx);
            sIdx[t][d] = yic * WW + xic;
            sWgt[t][d] = valid ? wv : 0.0f;
        }
    }
    __syncthreads();

    const float* xb = x + (size_t)b * C_IN * HW;

    // im2col assignment per chunk: 64 positions x 32 channels, 8 channels/thread
    const int my_p  = tid & 63;
    const int my_c0 = (tid >> 6) * 8;

    // sampling: fill sB[buf] for chunk ch (one ds_store_b128 per thread)
    auto sample_chunk = [&](int ch, int buf) {
        const int ktap = ch >> 3;
        const int c0   = (ch & 7) << 5;
        const int ci   = my_p * KTAPS + ktap;
        const int4   idx = *(const int4*)sIdx[ci];
        const float4 wgt = *(const float4*)sWgt[ci];
        const float* plane = xb + (size_t)(c0 + my_c0) * HW;
        float v[8];
#pragma unroll
        for (int j = 0; j < 8; ++j) {
            const float* pl = plane + (size_t)j * HW;
            v[j] = wgt.x * pl[idx.x] + wgt.y * pl[idx.y] +
                   wgt.z * pl[idx.z] + wgt.w * pl[idx.w];
        }
        uint4 packed;
        packed.x = pack_bf16x2(v[0], v[1]);
        packed.y = pack_bf16x2(v[2], v[3]);
        packed.z = pack_bf16x2(v[4], v[5]);
        packed.w = pack_bf16x2(v[6], v[7]);
        *(uint4*)&sB[buf][my_p * BROW + my_c0] = packed;
    };

    // ---- prologue: stage chunk 0 ----
    sample_chunk(0, 0);
    __syncthreads();

    v8f acc[2][4] = {};                       // [o-tile][n-tile]
    const int o0     = wave * 32;             // wave owns o-tiles o0, o0+16
    const int p_lane = lane & 15;
    const int hiHalf = (lane & 16) ? 1 : 0;   // K-half select (this code's contiguous k = 16h + s order)

    for (int ch = 0; ch < NCHUNK; ++ch) {
        const int cur  = ch & 1;
        const int ktap = ch >> 3;
        const int c0   = (ch & 7) << 5;
        const int K0   = ktap * 256 + c0;

        // B fragments: n-tile t = rows 16t..16t+15
        union { uint4 u[2]; v16bf v; } bf[4];
        const unsigned short* brow = &sB[cur][p_lane * BROW + hiHalf * 16];
#pragma unroll
        for (int t = 0; t < 4; ++t) {
            bf[t].u[0] = *(const uint4*)(brow + t * 16 * BROW);
            bf[t].u[1] = *(const uint4*)(brow + t * 16 * BROW + 8);
        }

        // A fragments: lane<16 -> row M=lane, K {0..15}; lane>=16 -> K {16..31}
        union { uint4 u[2]; v16bf v; } af[2];
        const unsigned short* arow = wbf + (size_t)(o0 + p_lane) * KDIM + K0 + hiHalf * 16;
        af[0].u[0] = *(const uint4*)(arow);
        af[0].u[1] = *(const uint4*)(arow + 8);
        af[1].u[0] = *(const uint4*)(arow + (size_t)16 * KDIM);
        af[1].u[1] = *(const uint4*)(arow + (size_t)16 * KDIM + 8);

        // stage next chunk into the other buffer (overlaps with WMMA below)
        if (ch + 1 < NCHUNK) {
            sample_chunk(ch + 1, cur ^ 1);
            const int ktap_n = (ch + 1) >> 3;
            const int K0n    = ktap_n * 256 + (((ch + 1) & 7) << 5);
            __builtin_prefetch(wbf + (size_t)(o0 + p_lane) * KDIM + K0n, 0, 1);
        }

#pragma unroll
        for (int t = 0; t < 4; ++t) {
            acc[0][t] = __builtin_amdgcn_wmma_f32_16x16x32_bf16(false, af[0].v, false, bf[t].v,
                                                               (short)0, acc[0][t], false, false);
            acc[1][t] = __builtin_amdgcn_wmma_f32_16x16x32_bf16(false, af[1].v, false, bf[t].v,
                                                               (short)0, acc[1][t], false, false);
        }

        __syncthreads();   // one barrier per chunk (double buffering)
    }

    // ---- ReLU + store. C layout: VGPR j, lane<16 -> (M=j, N=lane); lane>=16 -> (M=j+8, N=lane-16)
    const int mOff = hiHalf ? 8 : 0;
    float* outb = out + (size_t)b * C_OUT * PP + pbase + p_lane;
#pragma unroll
    for (int j = 0; j < 8; ++j) {
        const int m = j + mOff;
#pragma unroll
        for (int t = 0; t < 4; ++t) {
            outb[(size_t)(o0 + m)      * PP + 16 * t] = fmaxf(acc[0][t][j], 0.0f);
            outb[(size_t)(o0 + 16 + m) * PP + 16 * t] = fmaxf(acc[1][t][j], 0.0f);
        }
    }
}

extern "C" void kernel_launch(void* const* d_in, const int* in_sizes, int n_in,
                              void* d_out, int out_size, void* d_ws, size_t ws_size,
                              hipStream_t stream) {
    const float* x      = (const float*)d_in[0];
    const float* bboxes = (const float*)d_in[1];
    const float* scores = (const float*)d_in[2];
    const float* weight = (const float*)d_in[3];
    // d_in[4] = stride (==8), folded into constants
    unsigned short* wbf = (unsigned short*)d_ws;   // 256*2304*2 = 1.15 MB scratch

    const int wElems = C_OUT * KDIM;
    roidcn_repack_w<<<(wElems + 255) / 256, 256, 0, stream>>>(weight, wbf);

    // grid: B(2) * (16384/64 = 256) tiles
    roidcn_main<<<512, 256, 0, stream>>>(x, bboxes, scores, wbf, (float*)d_out);
}
